// GAT_29386166239459
// MI455X (gfx1250) — matrix operand order repaired and checked by reading, not compiled, not run on hardware
//
#include <hip/hip_runtime.h>
#include <hip/hip_bf16.h>

typedef __attribute__((ext_vector_type(16))) __bf16 v16bf;
typedef __attribute__((ext_vector_type(8)))  __bf16 v8bf;
typedef __attribute__((ext_vector_type(8)))  float  v8f;
typedef __attribute__((ext_vector_type(4)))  int    v4i;

#define NEG_SLOPE 0.2f
// encoded value of -inf for ordered-uint float max trick
#define ENC_NEG_INF 0x007FFFFFu

#if defined(__HIP_DEVICE_COMPILE__) && __has_builtin(__builtin_amdgcn_global_load_async_to_lds_b128)
#define HAVE_ASYNC_LDS 1
typedef __attribute__((address_space(1))) v4i* gv4i_p;   // global AS pointer to v4i
typedef __attribute__((address_space(3))) v4i* lv4i_p;   // LDS AS pointer to v4i
#endif

__device__ __forceinline__ unsigned enc_f32(float f) {
  unsigned u = __float_as_uint(f);
  return (u & 0x80000000u) ? ~u : (u | 0x80000000u);
}
__device__ __forceinline__ float dec_f32(unsigned k) {
  unsigned u = (k & 0x80000000u) ? (k & 0x7FFFFFFFu) : ~k;
  return __uint_as_float(u);
}
__device__ __forceinline__ float leaky(float x) { return x > 0.f ? x : NEG_SLOPE * x; }

// ---------------- conversion / prep ----------------

__global__ void k_cvt_bf16(const float* __restrict__ in, __bf16* __restrict__ out, long n) {
  long i = (long)blockIdx.x * blockDim.x + threadIdx.x;
  if (i < n) out[i] = (__bf16)in[i];
}

// W1 [128,128] -> W1t bf16 [col][k]; W2 [128,40] -> W2t bf16 padded [48][128]; b2 -> b2p[48]
__global__ void k_prep_w(const float* __restrict__ W1, const float* __restrict__ W2,
                         const float* __restrict__ b2,
                         __bf16* __restrict__ W1t, __bf16* __restrict__ W2t,
                         float* __restrict__ b2p) {
  int i = blockIdx.x * blockDim.x + threadIdx.x;
  if (i < 128 * 128) { int k = i >> 7, c = i & 127; W1t[c * 128 + k] = (__bf16)W1[i]; }
  if (i < 48 * 128)  { int c = i >> 7, k = i & 127; W2t[i] = (__bf16)(c < 40 ? W2[k * 40 + c] : 0.f); }
  if (i < 48)        { b2p[i] = (i < 40) ? b2[i] : 0.f; }
}

// ---------------- WMMA GEMM: C[M,ncols] = A[M,128] @ Bt^T + bias ----------------
// Bt is [ncols][128] bf16 (pre-transposed). blockDim.x = 32*(ncols/16).
__global__ void k_gemm_wmma(const __bf16* __restrict__ A, const __bf16* __restrict__ Bt,
                            const float* __restrict__ bias, float* __restrict__ C,
                            int M, int ncols) {
  __shared__ __attribute__((aligned(32))) __bf16 sA[16 * 128];
  const int tid = threadIdx.x, wave = tid >> 5, lane = tid & 31;
  const int row0 = blockIdx.x * 16;

  // stage A tile 16x128 bf16 (256 x 16B chunks), coalesced; prefer async copy to LDS
  {
    const int last = M - 1;
    for (int i = tid; i < 256; i += blockDim.x) {
      int srow = row0 + (i >> 4); if (srow > last) srow = last;
      const uint4* gp = ((const uint4*)(A + (size_t)srow * 128)) + (i & 15);
      uint4* lp = ((uint4*)sA) + i;
#ifdef HAVE_ASYNC_LDS
      __builtin_amdgcn_global_load_async_to_lds_b128((gv4i_p)gp, (lv4i_p)lp, 0, 0);
#else
      *lp = *gp;
#endif
    }
  }

  const int col0 = wave * 16;
  const int r = lane & 15, kg = lane >> 4;
  const __bf16* bbase = Bt + (size_t)(col0 + r) * 128;
  __builtin_prefetch(bbase, 0, 3);   // global_prefetch_b8 on the B panel

#ifdef HAVE_ASYNC_LDS
#if __has_builtin(__builtin_amdgcn_s_wait_asynccnt)
  __builtin_amdgcn_s_wait_asynccnt(0);
#else
  asm volatile("s_wait_asynccnt 0x0" ::: "memory");
#endif
#endif
  __syncthreads();

  v8f acc = {0.f, 0.f, 0.f, 0.f, 0.f, 0.f, 0.f, 0.f};
#pragma unroll
  for (int kk = 0; kk < 128; kk += 32) {
    // A fragment: lanes 0-15 hold row r, K={kk..kk+15};
    //             lanes 16-31 hold row r, K={kk+16..kk+31}
    const __bf16* ap = sA + r * 128 + kk + kg * 16;
    v8bf alo = *(const v8bf*)ap;
    v8bf ahi = *(const v8bf*)(ap + 8);
    v16bf a;
#pragma unroll
    for (int i = 0; i < 8; ++i) { a[i] = alo[i]; a[i + 8] = ahi[i]; }
    // B fragment: lane holds column col0+r, K = kk + kg*16 + {0..15} (contiguous)
    v16bf b = *(const v16bf*)(bbase + kk + kg * 16);
    acc = __builtin_amdgcn_wmma_f32_16x16x32_bf16(false, a, false, b, (short)0, acc,
                                                  false, false);
  }
  // D layout: VGPR v -> row v (lanes 0-15) / row v+8 (lanes 16-31), col = lane&15
  const int ccol = col0 + r;
  const float bc = bias[ccol];
  float* cp = C + (size_t)(row0 + kg * 8) * ncols + ccol;
  if (row0 + 16 <= M) {               // full tile: branch-free strided stores
#pragma unroll
    for (int v = 0; v < 8; ++v) { *cp = acc[v] + bc; cp += ncols; }
  } else {                            // ragged last tile
    int rr = row0 + kg * 8;
#pragma unroll
    for (int v = 0; v < 8; ++v) { if (rr + v < M) *cp = acc[v] + bc; cp += ncols; }
  }
}

// ---------------- layer-1 node prep: logits + init accumulators ----------------

__global__ void k_node1(const float* __restrict__ h1, const float* __restrict__ as1,
                        const float* __restrict__ ad1, float* __restrict__ es,
                        float* __restrict__ ed, unsigned* __restrict__ mxk,
                        float* __restrict__ den, float* __restrict__ agg, int Nn) {
  int i = blockIdx.x * blockDim.x + threadIdx.x;  // n*8 + h
  if (i >= Nn * 8) return;
  int n = i >> 3, h = i & 7;
  const float* hp = h1 + (size_t)n * 128 + h * 16;
  float s = 0.f, d = 0.f;
#pragma unroll
  for (int c = 0; c < 16; ++c) { s += hp[c] * as1[h * 16 + c]; d += hp[c] * ad1[h * 16 + c]; }
  es[i] = s; ed[i] = d;
  mxk[i] = ENC_NEG_INF; den[i] = 0.f;
  float* ap = agg + (size_t)n * 128 + h * 16;
#pragma unroll
  for (int c = 0; c < 16; ++c) ap[c] = 0.f;
}

// ---------------- layer-1 edge passes (index >= E are self loops) ----------------

__global__ void k_emax1(const long long* __restrict__ src, const long long* __restrict__ dst,
                        const float* __restrict__ es, const float* __restrict__ ed,
                        unsigned* __restrict__ mxk, long long E, int Nn) {
  long long e = (long long)blockIdx.x * blockDim.x + threadIdx.x;
  if (e >= E + Nn) return;
  long long s = (e < E) ? src[e] : (e - E);
  long long d = (e < E) ? dst[e] : (e - E);
#pragma unroll
  for (int h = 0; h < 8; ++h) {
    float a = leaky(es[s * 8 + h] + ed[d * 8 + h]);
    atomicMax(&mxk[d * 8 + h], enc_f32(a));
  }
}

__global__ void k_esum1(const long long* __restrict__ src, const long long* __restrict__ dst,
                        const float* __restrict__ es, const float* __restrict__ ed,
                        const unsigned* __restrict__ mxk, float* __restrict__ den,
                        long long E, int Nn) {
  long long e = (long long)blockIdx.x * blockDim.x + threadIdx.x;
  if (e >= E + Nn) return;
  long long s = (e < E) ? src[e] : (e - E);
  long long d = (e < E) ? dst[e] : (e - E);
#pragma unroll
  for (int h = 0; h < 8; ++h) {
    float a = leaky(es[s * 8 + h] + ed[d * 8 + h]);
    atomicAdd(&den[d * 8 + h], __expf(a - dec_f32(mxk[d * 8 + h])));
  }
}

// one wave per edge; lane covers 4 channels (all within one head since 4|16)
__global__ void k_eagg1(const long long* __restrict__ src, const long long* __restrict__ dst,
                        const float* __restrict__ es, const float* __restrict__ ed,
                        const unsigned* __restrict__ mxk, const float* __restrict__ den,
                        const float* __restrict__ h1, float* __restrict__ agg,
                        long long E, int Nn) {
  long long idx = (long long)blockIdx.x * blockDim.x + threadIdx.x;
  long long e = idx >> 5; int lane = (int)(idx & 31);
  if (e >= E + Nn) return;
  long long s = (e < E) ? src[e] : (e - E);
  long long d = (e < E) ? dst[e] : (e - E);
  int h = lane >> 2;
  float a = leaky(es[s * 8 + h] + ed[d * 8 + h]);
  float wgt = __expf(a - dec_f32(mxk[d * 8 + h])) / (den[d * 8 + h] + 1e-16f);
  float4 hv = ((const float4*)(h1 + (size_t)s * 128))[lane];
  float* o = agg + (size_t)d * 128 + lane * 4;
  atomicAdd(o + 0, hv.x * wgt);
  atomicAdd(o + 1, hv.y * wgt);
  atomicAdd(o + 2, hv.z * wgt);
  atomicAdd(o + 3, hv.w * wgt);
}

// relu(agg + b1) -> bf16 input for GEMM2
__global__ void k_relu_cvt(const float* __restrict__ agg, const float* __restrict__ b1,
                           __bf16* __restrict__ h1b, long n) {
  long i = (long)blockIdx.x * blockDim.x + threadIdx.x;
  if (i >= n) return;
  float v = agg[i] + b1[i & 127];
  h1b[i] = (__bf16)(v > 0.f ? v : 0.f);
}

// ---------------- layer-2 node prep ----------------

__global__ void k_node2(const float* __restrict__ h2 /*[N,48]*/, const float* __restrict__ as2,
                        const float* __restrict__ ad2, float* __restrict__ es2,
                        float* __restrict__ ed2, unsigned* __restrict__ mxk2,
                        float* __restrict__ den2, float* __restrict__ agg2, int Nn) {
  int n = blockIdx.x * blockDim.x + threadIdx.x;
  if (n >= Nn) return;
  const float* hp = h2 + (size_t)n * 48;
  float s = 0.f, d = 0.f;
#pragma unroll
  for (int c = 0; c < 40; ++c) { s += hp[c] * as2[c]; d += hp[c] * ad2[c]; }
  es2[n] = s; ed2[n] = d;
  mxk2[n] = ENC_NEG_INF; den2[n] = 0.f;
  float* ap = agg2 + (size_t)n * 40;
#pragma unroll
  for (int c = 0; c < 40; ++c) ap[c] = 0.f;
}

// ---------------- layer-2 edge passes (1 head, 40 channels) ----------------

__global__ void k_emax2(const long long* __restrict__ src, const long long* __restrict__ dst,
                        const float* __restrict__ es2, const float* __restrict__ ed2,
                        unsigned* __restrict__ mxk2, long long E, int Nn) {
  long long e = (long long)blockIdx.x * blockDim.x + threadIdx.x;
  if (e >= E + Nn) return;
  long long s = (e < E) ? src[e] : (e - E);
  long long d = (e < E) ? dst[e] : (e - E);
  atomicMax(&mxk2[d], enc_f32(leaky(es2[s] + ed2[d])));
}

__global__ void k_esum2(const long long* __restrict__ src, const long long* __restrict__ dst,
                        const float* __restrict__ es2, const float* __restrict__ ed2,
                        const unsigned* __restrict__ mxk2, float* __restrict__ den2,
                        long long E, int Nn) {
  long long e = (long long)blockIdx.x * blockDim.x + threadIdx.x;
  if (e >= E + Nn) return;
  long long s = (e < E) ? src[e] : (e - E);
  long long d = (e < E) ? dst[e] : (e - E);
  float a = leaky(es2[s] + ed2[d]);
  atomicAdd(&den2[d], __expf(a - dec_f32(mxk2[d])));
}

// one wave per edge; lane covers col lane and (lane+32 if < 40)
__global__ void k_eagg2(const long long* __restrict__ src, const long long* __restrict__ dst,
                        const float* __restrict__ es2, const float* __restrict__ ed2,
                        const unsigned* __restrict__ mxk2, const float* __restrict__ den2,
                        const float* __restrict__ h2, float* __restrict__ agg2,
                        long long E, int Nn) {
  long long idx = (long long)blockIdx.x * blockDim.x + threadIdx.x;
  long long e = idx >> 5; int lane = (int)(idx & 31);
  if (e >= E + Nn) return;
  long long s = (e < E) ? src[e] : (e - E);
  long long d = (e < E) ? dst[e] : (e - E);
  float a = leaky(es2[s] + ed2[d]);
  float wgt = __expf(a - dec_f32(mxk2[d])) / (den2[d] + 1e-16f);
  const float* hp = h2 + (size_t)s * 48;
  float* o = agg2 + (size_t)d * 40;
  atomicAdd(&o[lane], hp[lane] * wgt);
  if (lane < 8) atomicAdd(&o[lane + 32], hp[lane + 32] * wgt);
}

// ---------------- bias + log_softmax over 40 classes ----------------

__global__ void k_final(const float* __restrict__ agg2, const float* __restrict__ b2,
                        float* __restrict__ out, int Nn) {
  int n = blockIdx.x * blockDim.x + threadIdx.x;
  if (n >= Nn) return;
  float v[40];
  float m = -3.4e38f;
#pragma unroll
  for (int c = 0; c < 40; ++c) { v[c] = agg2[(size_t)n * 40 + c] + b2[c]; m = fmaxf(m, v[c]); }
  float sum = 0.f;
#pragma unroll
  for (int c = 0; c < 40; ++c) sum += __expf(v[c] - m);
  float lse = __logf(sum) + m;
#pragma unroll
  for (int c = 0; c < 40; ++c) out[(size_t)n * 40 + c] = v[c] - lse;
}

// ---------------- host launcher ----------------

extern "C" void kernel_launch(void* const* d_in, const int* in_sizes, int n_in,
                              void* d_out, int out_size, void* d_ws, size_t ws_size,
                              hipStream_t stream) {
  const float*     x   = (const float*)d_in[0];
  const long long* ei  = (const long long*)d_in[1];   // int64 edge_index [2,E]
  const float*     W1  = (const float*)d_in[2];
  const float*     as1 = (const float*)d_in[3];
  const float*     ad1 = (const float*)d_in[4];
  const float*     b1  = (const float*)d_in[5];
  const float*     W2  = (const float*)d_in[6];
  const float*     as2 = (const float*)d_in[7];
  const float*     ad2 = (const float*)d_in[8];
  const float*     b2  = (const float*)d_in[9];

  const int       Nn  = in_sizes[0] / 128;
  const long long E   = in_sizes[1] / 2;
  const long long* srcI = ei;
  const long long* dstI = ei + E;
  float* out = (float*)d_out;

  // carve workspace (256B-aligned slices)
  char* wsb = (char*)d_ws; size_t off = 0;
  auto take = [&](size_t bytes) -> void* {
    void* p = wsb + off; off += (bytes + 255) & ~(size_t)255; return p;
  };
  __bf16*   xb   = (__bf16*)take((size_t)Nn * 128 * 2);
  __bf16*   W1t  = (__bf16*)take(128 * 128 * 2);
  __bf16*   W2t  = (__bf16*)take(48 * 128 * 2);
  float*    b2p  = (float*)take(48 * 4);
  float*    h1   = (float*)take((size_t)Nn * 128 * 4);
  float*    es1  = (float*)take((size_t)Nn * 8 * 4);
  float*    ed1  = (float*)take((size_t)Nn * 8 * 4);
  unsigned* mxk1 = (unsigned*)take((size_t)Nn * 8 * 4);
  float*    den1 = (float*)take((size_t)Nn * 8 * 4);
  float*    agg1 = (float*)take((size_t)Nn * 128 * 4);
  __bf16*   h1b  = (__bf16*)take((size_t)Nn * 128 * 2);
  float*    h2   = (float*)take((size_t)Nn * 48 * 4);
  float*    es2  = (float*)take((size_t)Nn * 4);
  float*    ed2  = (float*)take((size_t)Nn * 4);
  unsigned* mxk2 = (unsigned*)take((size_t)Nn * 4);
  float*    den2 = (float*)take((size_t)Nn * 4);
  float*    agg2 = (float*)take((size_t)Nn * 40 * 4);

  const int T = 256;
  const long nx = (long)Nn * 128;
  const long long ET = E + Nn;                 // edges + self loops
  const unsigned ebl  = (unsigned)((ET + T - 1) / T);
  const unsigned wbl  = (unsigned)((ET * 32 + T - 1) / T);   // wave-per-edge kernels
  const unsigned m16  = (unsigned)((Nn + 15) / 16);

  // prep: bf16 conversions, transposed/padded weights
  k_cvt_bf16<<<(unsigned)((nx + T - 1) / T), T, 0, stream>>>(x, xb, nx);
  k_prep_w<<<64, T, 0, stream>>>(W1, W2, b2, W1t, W2t, b2p);

  // layer 1
  k_gemm_wmma<<<m16, 256, 0, stream>>>(xb, W1t, b1, h1, Nn, 128);
  k_node1<<<(unsigned)((Nn * 8 + T - 1) / T), T, 0, stream>>>(h1, as1, ad1, es1, ed1,
                                                              mxk1, den1, agg1, Nn);
  k_emax1<<<ebl, T, 0, stream>>>(srcI, dstI, es1, ed1, mxk1, E, Nn);
  k_esum1<<<ebl, T, 0, stream>>>(srcI, dstI, es1, ed1, mxk1, den1, E, Nn);
  k_eagg1<<<wbl, T, 0, stream>>>(srcI, dstI, es1, ed1, mxk1, den1, h1, agg1, E, Nn);
  k_relu_cvt<<<(unsigned)((nx + T - 1) / T), T, 0, stream>>>(agg1, b1, h1b, nx);

  // layer 2 (cols padded 40 -> 48 = 3 WMMA tiles; blockDim = 3 waves)
  k_gemm_wmma<<<m16, 96, 0, stream>>>(h1b, W2t, b2p, h2, Nn, 48);
  k_node2<<<(unsigned)((Nn + T - 1) / T), T, 0, stream>>>(h2, as2, ad2, es2, ed2,
                                                          mxk2, den2, agg2, Nn);
  k_emax2<<<ebl, T, 0, stream>>>(srcI, dstI, es2, ed2, mxk2, E, Nn);
  k_esum2<<<ebl, T, 0, stream>>>(srcI, dstI, es2, ed2, mxk2, den2, E, Nn);
  k_eagg2<<<wbl, T, 0, stream>>>(srcI, dstI, es2, ed2, mxk2, den2, h2, agg2, E, Nn);

  k_final<<<(unsigned)((Nn + T - 1) / T), T, 0, stream>>>(agg2, b2, out, Nn);
}
